// multihead_self_attention_78340203479353
// MI455X (gfx1250) — matrix operand order repaired and checked by reading, not compiled, not run on hardware
//
#include <hip/hip_runtime.h>

typedef __attribute__((ext_vector_type(16))) __bf16 v16bf;
typedef __attribute__((ext_vector_type(8)))  __bf16 v8bf;
typedef __attribute__((ext_vector_type(8)))  float  v8f;

#define BB 2
#define SS 2048
#define DD 1024
#define HH 16
#define DK 64
#define MM (BB*SS)   /* 4096 rows of flattened (b,s) */

__device__ __forceinline__ v8f wmma_bf16(v16bf a, v16bf b, v8f c) {
  // D = A(16x32 bf16) * B(32x16 bf16) + C(16x16 f32)
  return __builtin_amdgcn_wmma_f32_16x16x32_bf16(false, a, false, b, (short)0, c, false, false);
}

enum { MODE_PLAIN = 0, MODE_ROPE = 1, MODE_VTRANS = 2 };

// C[m,n] = sum_k A[m,k] * W[n,k]   (A: [MM,DD] row-major, W: [DD,DD] row-major)
// One wave -> 32x64 output tile (2 M-subtiles share B fragments). Block = 4 waves -> 128x64.
template<int MODE>
__global__ __launch_bounds__(128)
void gemm_proj(const float* __restrict__ A,
               const float* __restrict__ W,
               void* __restrict__ outp,
               float scale)
{
  const int lane = threadIdx.x & 31;
  const int wave = threadIdx.x >> 5;
  const int l15  = lane & 15;
  const int half = lane >> 4;
  const int m0 = blockIdx.x * 128 + wave * 32;
  const int n0 = blockIdx.y * 64;

  v8f acc[2][4] = {};
  const float* arow0 = A + (size_t)(m0 + l15) * DD;
  const float* arow1 = arow0 + (size_t)16 * DD;

  for (int kb = 0; kb < DD; kb += 32) {
    // A-fragments (16x32 bf16): lane row; elems: k = kb + (e>>3)*16 + half*8 + (e&7)
    v16bf af[2];
    {
      const float* ap0 = arow0 + kb + half * 8;
      const float* ap1 = arow1 + kb + half * 8;
      __builtin_prefetch(ap0 + 32, 0, 1);
      __builtin_prefetch(ap1 + 32, 0, 1);
#pragma unroll
      for (int i = 0; i < 8; ++i) {
        af[0][i]     = (__bf16)ap0[i];
        af[0][i + 8] = (__bf16)ap0[i + 16];
        af[1][i]     = (__bf16)ap1[i];
        af[1][i + 8] = (__bf16)ap1[i + 16];
      }
    }
#pragma unroll
    for (int sub = 0; sub < 4; ++sub) {
      // B-fragment (32x16 bf16): lane col n = n0+sub*16+l15; elems: k = kb + (e>>3)*16 + half*8 + (e&7)
      const float* wp = W + (size_t)(n0 + sub * 16 + l15) * DD + kb + half * 8;
      v16bf bfr;
#pragma unroll
      for (int i = 0; i < 8; ++i) { bfr[i] = (__bf16)wp[i]; bfr[i + 8] = (__bf16)wp[i + 16]; }
      acc[0][sub] = wmma_bf16(af[0], bfr, acc[0][sub]);
      acc[1][sub] = wmma_bf16(af[1], bfr, acc[1][sub]);
    }
  }

  // C layout: reg r holds (m = m0 + j*16 + half*8 + r, n = n0 + sub*16 + l15)
#pragma unroll
  for (int j = 0; j < 2; ++j) {
    const int mj = m0 + j * 16;
    if constexpr (MODE == MODE_PLAIN) {
      float* out = (float*)outp;  // [MM, DD]
#pragma unroll
      for (int sub = 0; sub < 4; ++sub) {
        int n = n0 + sub * 16 + l15;
#pragma unroll
        for (int r = 0; r < 8; ++r) {
          int m = mj + half * 8 + r;
          out[(size_t)m * DD + n] = acc[j][sub][r];
        }
      }
    } else if constexpr (MODE == MODE_ROPE) {
      // RoPE + head transpose; out bf16 [B,H,S,DK]
      __bf16* out = (__bf16*)outp;
#pragma unroll
      for (int sub = 0; sub < 4; ++sub) {
        int n = n0 + sub * 16 + l15;
        int d = n & (DK - 1);
        int h = n >> 6;
        // inv_freq = theta^(-2*(d>>1)/64) = exp(-ln(1e4)*(d>>1)/32)
        float inv_freq = __expf(-9.210340371976184f * (float)(d >> 1) * (1.0f / 32.0f));
#pragma unroll
        for (int r = 0; r < 8; ++r) {
          int m = mj + half * 8 + r;
          int s = m & (SS - 1);
          int b = m >> 11;
          float v = acc[j][sub][r] * scale;
          float p = __shfl_xor(v, 1, 32);        // partner of the (even,odd) pair
          float ang = (float)s * inv_freq;
          float sn, cs;
          __sincosf(ang, &sn, &cs);
          float y = (d & 1) ? (sn * p + cs * v)   // odd:  sin*x_even + cos*x_odd
                            : (cs * v - sn * p);  // even: cos*x_even - sin*x_odd
          out[(((size_t)(b * HH + h) * SS) + s) * DK + d] = (__bf16)y;
        }
      }
    } else {
      // V: bf16, stored d-major [B,H,DK,S] so attention B-fragments read contiguous
      __bf16* out = (__bf16*)outp;
      int b = mj >> 11;
      int sbase = (mj & (SS - 1)) + half * 8;
#pragma unroll
      for (int sub = 0; sub < 4; ++sub) {
        int n = n0 + sub * 16 + l15;
        int d = n & (DK - 1);
        int h = n >> 6;
        v8bf pk;
#pragma unroll
        for (int r = 0; r < 8; ++r) pk[r] = (__bf16)acc[j][sub][r];
        *(v8bf*)(out + (((size_t)(b * HH + h) * DK) + d) * SS + sbase) = pk;
      }
    }
  }
}

// Flash attention: one wave = 32 queries (two 16-row q-tiles) of one (b,h).
// Sᵀ = K·Qᵀ trick: the softmax'ed tile maps lane-locally onto the P·V A-fragment.
// K/V fragments are loaded once per 32-key step and reused by both q-tiles.
__global__ __launch_bounds__(128)
void attn_flash(const __bf16* __restrict__ Qb,  // [B,H,S,DK]  (pre-scaled by 1/sqrt(dk))
                const __bf16* __restrict__ Kb,  // [B,H,S,DK]
                const __bf16* __restrict__ Vt,  // [B,H,DK,S]
                float* __restrict__ AO)         // [B,S,H*DK]
{
  const int lane = threadIdx.x & 31;
  const int wave = threadIdx.x >> 5;
  const int l15  = lane & 15;
  const int half = lane >> 4;
  const int qp = blockIdx.x * 4 + wave;     // 0..63
  const int h  = blockIdx.y;
  const int b  = blockIdx.z;
  const int q0 = qp * 32;                   // 32-aligned

  const __bf16* Qh = Qb + (size_t)(b * HH + h) * SS * DK;
  const __bf16* Kh = Kb + (size_t)(b * HH + h) * SS * DK;
  const __bf16* Vh = Vt + (size_t)(b * HH + h) * DK * SS;

  // Q as B-fragment for Sᵀ = K·Qᵀ : lane col q = q0+j*16+l15, elems dd = g*32 + (e>>3)*16 + half*8 + (e&7)
  v16bf qf[2][2];
#pragma unroll
  for (int j = 0; j < 2; ++j)
#pragma unroll
    for (int g = 0; g < 2; ++g) {
      const __bf16* qp_ = Qh + (size_t)(q0 + j * 16 + l15) * DK + g * 32 + half * 8;
#pragma unroll
      for (int i = 0; i < 8; ++i) { qf[j][g][i] = qp_[i]; qf[j][g][i + 8] = qp_[i + 16]; }
    }

  v8f o[2][4] = {};              // O tiles 16q x 64d; reg r -> q = half*8+r, lane -> d
  float m_run[2] = {-1e30f, -1e30f};
  float l_run[2] = {0.0f, 0.0f};

  for (int k0 = 0; k0 < q0 + 32; k0 += 32) {
    // Sᵀ tiles: st[j][t] covers keys k0+t*16..+15 (M) x queries q0+j*16.. (N)
    v8f st[2][2] = {};
#pragma unroll
    for (int t = 0; t < 2; ++t) {
#pragma unroll
      for (int g = 0; g < 2; ++g) {
        // K as A-fragment: lane row key = k0+t*16+l15; elems dd = g*32+(e>>3)*16+half*8+(e&7)
        const __bf16* kp = Kh + (size_t)(k0 + t * 16 + l15) * DK + g * 32 + half * 8;
        v16bf kf;
#pragma unroll
        for (int i = 0; i < 8; ++i) { kf[i] = kp[i]; kf[i + 8] = kp[i + 16]; }
        st[0][t] = wmma_bf16(kf, qf[0][g], st[0][t]);
        st[1][t] = wmma_bf16(kf, qf[1][g], st[1][t]);
      }
    }

    v16bf pf[2];
#pragma unroll
    for (int j = 0; j < 2; ++j) {
      // causal mask on boundary tiles (lane: query q0+j*16+l15; reg: key k0+t*16+half*8+r)
      if (k0 + 31 > q0 + j * 16) {
        int qa = q0 + j * 16 + l15;
#pragma unroll
        for (int t = 0; t < 2; ++t)
#pragma unroll
          for (int r = 0; r < 8; ++r) {
            int ka = k0 + t * 16 + half * 8 + r;
            if (ka > qa) st[j][t][r] = -1e30f;
          }
      }

      // per-query max: 16 in-register values, then fold halves
      float mx = st[j][0][0];
#pragma unroll
      for (int r = 1; r < 8; ++r) mx = fmaxf(mx, st[j][0][r]);
#pragma unroll
      for (int r = 0; r < 8; ++r) mx = fmaxf(mx, st[j][1][r]);
      mx = fmaxf(mx, __shfl_xor(mx, 16, 32));
      float m_new = fmaxf(m_run[j], mx);
      float corr = __expf(m_run[j] - m_new);

      // P fragment (A layout 16x32): elem e = exp(st[e>>3][e&7] - m), lane-local.
      float psum = 0.0f;
#pragma unroll
      for (int e = 0; e < 16; ++e) {
        float p = __expf(st[j][e >> 3][e & 7] - m_new);
        psum += p;
        pf[j][e] = (__bf16)p;
      }
      psum += __shfl_xor(psum, 16, 32);
      l_run[j] = l_run[j] * corr + psum;
      m_run[j] = m_new;

      // rescale O: need corr of query (half*8+r) -> fetch from lane holding that query
      float cr[8];
#pragma unroll
      for (int r = 0; r < 8; ++r) cr[r] = __shfl(corr, r + half * 8, 32);
#pragma unroll
      for (int sub = 0; sub < 4; ++sub)
#pragma unroll
        for (int r = 0; r < 8; ++r) o[j][sub][r] *= cr[r];
    }

    // O += P(16x32) · V(32x64): V B-fragment lane col d = sub*16+l15, elems k = (e>>3)*16+half*8+(e&7)
#pragma unroll
    for (int sub = 0; sub < 4; ++sub) {
      const __bf16* vp = Vh + (size_t)(sub * 16 + l15) * SS + k0 + half * 8;
      v16bf vf;
#pragma unroll
      for (int i = 0; i < 8; ++i) { vf[i] = vp[i]; vf[i + 8] = vp[i + 16]; }
      o[0][sub] = wmma_bf16(pf[0], vf, o[0][sub]);
      o[1][sub] = wmma_bf16(pf[1], vf, o[1][sub]);
    }
  }

  // normalize by l and store to [B,S,H*DK] f32
#pragma unroll
  for (int j = 0; j < 2; ++j) {
    float li[8];
#pragma unroll
    for (int r = 0; r < 8; ++r) {
      float lr = __shfl(l_run[j], r + half * 8, 32);
      li[r] = 1.0f / lr;
    }
#pragma unroll
    for (int sub = 0; sub < 4; ++sub) {
      int d = sub * 16 + l15;
#pragma unroll
      for (int r = 0; r < 8; ++r) {
        int q = q0 + j * 16 + half * 8 + r;
        AO[((size_t)(b * SS + q)) * DD + h * DK + d] = o[j][sub][r] * li[r];
      }
    }
  }
}

extern "C" void kernel_launch(void* const* d_in, const int* in_sizes, int n_in,
                              void* d_out, int out_size, void* d_ws, size_t ws_size,
                              hipStream_t stream) {
  const float* X  = (const float*)d_in[0];
  const float* WQ = (const float*)d_in[1];
  const float* WK = (const float*)d_in[2];
  const float* WV = (const float*)d_in[3];
  const float* WO = (const float*)d_in[4];

  // workspace: Qb(8MB) Kb(8MB) Vt(8MB) bf16, AO(16MB) f32  => 40MB
  __bf16* Qb = (__bf16*)d_ws;
  __bf16* Kb = Qb + (size_t)MM * DD;
  __bf16* Vt = Kb + (size_t)MM * DD;
  float*  AO = (float*)(Vt + (size_t)MM * DD);

  dim3 gblk(128);
  dim3 ggrid(MM / 128, DD / 64);

  const float qscale = 0.125f;  // 1/sqrt(64)
  gemm_proj<MODE_ROPE>  <<<ggrid, gblk, 0, stream>>>(X, WQ, (void*)Qb, qscale);
  gemm_proj<MODE_ROPE>  <<<ggrid, gblk, 0, stream>>>(X, WK, (void*)Kb, 1.0f);
  gemm_proj<MODE_VTRANS><<<ggrid, gblk, 0, stream>>>(X, WV, (void*)Vt, 1.0f);

  attn_flash<<<dim3(SS / 32 / 4, HH, BB), gblk, 0, stream>>>(Qb, Kb, Vt, AO);

  gemm_proj<MODE_PLAIN> <<<ggrid, gblk, 0, stream>>>(AO, WO, d_out, 1.0f);
}
